// MSTRBlock_63943473102979
// MI455X (gfx1250) — matrix operand order repaired and checked by reading, not compiled, not run on hardware
//
#include <hip/hip_runtime.h>
#include <hip/hip_bf16.h>
#include <math.h>

// ---------------------------------------------------------------------------
// MI455X (gfx1250): bf16 WMMA GEMMs fed by the Tensor Data Mover.
//  - TDM double-buffers 128x64 A / 64x64 B K-slabs into LDS (pad -> 144B rows)
//  - wave0 issues tensor_load_to_lds, s_wait_tensorcnt 0, then barrier
//  - v_wmma_f32_16x16x32_bf16, fp32 accum; fp32 LN/softmax/GELU epilogues
// ---------------------------------------------------------------------------

typedef __bf16 bf16;
typedef __attribute__((ext_vector_type(16))) __bf16 v16bf;
typedef __attribute__((ext_vector_type(8)))  __bf16 v8bf;
typedef __attribute__((ext_vector_type(8)))  float  v8f;
typedef __attribute__((ext_vector_type(4)))  unsigned int v4u;
typedef __attribute__((ext_vector_type(8)))  int v8i;
typedef __attribute__((ext_vector_type(4)))  int v4i;

#define DD 768
#define HH 6
#define HD 128
#define BB 8
#define TT 4096

#define KS       64            // K-slab (bf16 elems) staged per TDM transfer
#define ROWB     144           // LDS row pitch: 128B data + 16B pad
#define A_ROWS   128
#define B_ROWS   64
#define TILE_B   ((A_ROWS + B_ROWS) * ROWB)   // 27648 B per buffer

// ---------------------------------------------------------------------------
__global__ void cvt_f32_bf16(const float* __restrict__ in, bf16* __restrict__ out, long n) {
    long i = (long)blockIdx.x * blockDim.x + threadIdx.x;
    if (i < n) out[i] = (bf16)in[i];
}

__global__ void zero_f32(float* __restrict__ p, long n) {
    long i = (long)blockIdx.x * blockDim.x + threadIdx.x;
    if (i < n) p[i] = 0.f;
}

// ---------------------------------------------------------------------------
__global__ void __launch_bounds__(256) ln_kernel(const float* __restrict__ x,
                                                 const float* __restrict__ g,
                                                 const float* __restrict__ bta,
                                                 bf16* __restrict__ xn) {
    const int row = blockIdx.x;
    const float* xr = x + (size_t)row * DD;
    float s = 0.f, ss = 0.f;
    for (int i = threadIdx.x; i < DD; i += 256) { float v = xr[i]; s += v; ss += v * v; }
    for (int o = 16; o > 0; o >>= 1) { s += __shfl_xor(s, o, 32); ss += __shfl_xor(ss, o, 32); }
    __shared__ float sh[16];
    int wid = threadIdx.x >> 5, lane = threadIdx.x & 31;
    if (lane == 0) { sh[wid] = s; sh[8 + wid] = ss; }
    __syncthreads();
    if (threadIdx.x == 0) {
        float ts = 0.f, tss = 0.f;
        for (int w = 0; w < 8; ++w) { ts += sh[w]; tss += sh[8 + w]; }
        float mu = ts / DD;
        float var = tss / DD - mu * mu;
        sh[0] = mu; sh[1] = rsqrtf(var + 1e-5f);
    }
    __syncthreads();
    float mu = sh[0], rs = sh[1];
    bf16* xo = xn + (size_t)row * DD;
    for (int i = threadIdx.x; i < DD; i += 256)
        xo[i] = (bf16)((xr[i] - mu) * rs * g[i] + bta[i]);
}

// ---------------------------------------------------------------------------
__global__ void pool3_kernel(const bf16* __restrict__ in, bf16* __restrict__ out,
                             int inStride, int outLp) {
    long i = (long)blockIdx.x * blockDim.x + threadIdx.x;
    long total = (long)BB * outLp * DD;
    if (i >= total) return;
    int d = (int)(i % DD);
    long t = i / DD;
    int tl = (int)(t % outLp);
    int b  = (int)(t / outLp);
    const bf16* src = in + (size_t)b * inStride * DD + (size_t)(3 * tl) * DD + d;
    float v = ((float)src[0] + (float)src[DD] + (float)src[2 * DD]) * (1.f / 3.f);
    out[i] = (bf16)v;
}

// ---------------------------------------------------------------------------
// TDM: stage a 2-D tile (tileRows x 64 bf16) into LDS, padded rows (144 B).
// D# built per ISA ch.8: group0 {count,lds,global,type=2}; group1 {flags,dims}.
__device__ __forceinline__ void tdm_load_2d(unsigned ldsAddr, const bf16* gptr,
                                            unsigned remRows, unsigned remK,
                                            unsigned strideElems, unsigned tileRows) {
    unsigned long long ga = (unsigned long long)(size_t)gptr;
    v4u g0;
    g0[0] = 1u;                                         // count=1, user mode
    g0[1] = ldsAddr;                                    // lds_addr
    g0[2] = (unsigned)(ga & 0xFFFFFFFFu);               // global_addr[31:0]
    g0[3] = (unsigned)((ga >> 32) & 0x01FFFFFFu)        // global_addr[56:32]
          | (2u << 30);                                 // type=2 (image)
    // flags: data_size=1 (2B) | pad_enable | pad_interval=4 (32 dw) | pad_amount=3 (4 dw)
    unsigned fl = (1u << 16) | (1u << 20) | (4u << 22) | (3u << 25);
    v8i g1;
    g1[0] = (int)fl;
    g1[1] = (int)((remK & 0xFFFFu) << 16);                                  // tdim0[15:0]
    g1[2] = (int)(((remK >> 16) & 0xFFFFu) | ((remRows & 0xFFFFu) << 16));  // tdim0[31:16]|tdim1[15:0]
    g1[3] = (int)(((remRows >> 16) & 0xFFFFu) | ((unsigned)KS << 16));      // tdim1[31:16]|tile_dim0
    g1[4] = (int)(tileRows & 0xFFFFu);                                      // tile_dim1 (tile_dim2=0)
    g1[5] = (int)strideElems;                                               // tensor_dim0_stride[31:0]
    g1[6] = 0;
    g1[7] = 0;
    v4i z4 = {0, 0, 0, 0};
    v8i z8 = {0, 0, 0, 0, 0, 0, 0, 0};
    __builtin_amdgcn_tensor_load_to_lds(g0, g1, z4, z4, z8, 0);
}

// ---------------------------------------------------------------------------
// C(M,N) = A(M,K)bf16 @ W(N,K)bf16^T + bias. Block 256 thr, tile 128x64,
// TDM double-buffered LDS staging, WMMA 16x16x32 bf16 from ds_load_b128.
// OMODE: 0 = bf16 store, 1 = f32 store, 2 = f32 store + residual add
template <int OMODE>
__global__ void __launch_bounds__(256)
gemm_wmma(const bf16* __restrict__ A, const bf16* __restrict__ Wm,
          const float* __restrict__ bias, void* __restrict__ Cout,
          const float* __restrict__ resid, int M, int N, int K) {
    __shared__ __align__(32) char lds[2][TILE_B];

    const int bm = blockIdx.x * 128;
    const int bn = blockIdx.y * 64;
    const int wave = threadIdx.x >> 5;
    const int lane = threadIdx.x & 31;
    const int wm = wave & 3, wn = wave >> 2;
    const int l15 = lane & 15;
    const int kbA = (lane < 16) ? 0 : 16;  // A frag: 16 contiguous K per lane
    const int kbB = (lane < 16) ? 0 : 16;  // B frag: 16 contiguous K per lane

    const unsigned remRowsA = (unsigned)(M - bm);
    const unsigned remRowsB = (unsigned)(N - bn);
    const int nslab = K / KS;
    const bool issuer = (wave == 0);

    // prologue: slab 0 -> buffer 0
    if (issuer) {
        unsigned lb = (unsigned)(size_t)&lds[0][0];
        tdm_load_2d(lb,                 A  + (size_t)bm * K, remRowsA, (unsigned)K, (unsigned)K, A_ROWS);
        tdm_load_2d(lb + A_ROWS * ROWB, Wm + (size_t)bn * K, remRowsB, (unsigned)K, (unsigned)K, B_ROWS);
        __builtin_amdgcn_s_wait_tensorcnt(0);
    }
    __syncthreads();

    v8f acc[2][2] = {};
    for (int s = 0; s < nslab; ++s) {
        if (s + 1 < nslab && issuer) {   // overlap: DMA next slab into other buffer
            int k0 = (s + 1) * KS;
            unsigned lb = (unsigned)(size_t)&lds[(s + 1) & 1][0];
            tdm_load_2d(lb,                 A  + (size_t)bm * K + k0, remRowsA, (unsigned)(K - k0), (unsigned)K, A_ROWS);
            tdm_load_2d(lb + A_ROWS * ROWB, Wm + (size_t)bn * K + k0, remRowsB, (unsigned)(K - k0), (unsigned)K, B_ROWS);
        }
        const char* bufA = &lds[s & 1][0];
        const char* bufB = &lds[s & 1][A_ROWS * ROWB];
#pragma unroll
        for (int ks = 0; ks < KS; ks += 32) {
            v16bf af[2], bfrag[2];
#pragma unroll
            for (int rt = 0; rt < 2; ++rt) {
                const char* p = bufA + (wm * 32 + rt * 16 + l15) * ROWB + (ks + kbA) * 2;
                ((v8bf*)&af[rt])[0] = *(const v8bf*)p;
                ((v8bf*)&af[rt])[1] = *(const v8bf*)(p + 16);   // next 8 K elems
            }
#pragma unroll
            for (int ct = 0; ct < 2; ++ct) {
                const char* p = bufB + (wn * 32 + ct * 16 + l15) * ROWB + (ks + kbB) * 2;
                ((v8bf*)&bfrag[ct])[0] = *(const v8bf*)p;
                ((v8bf*)&bfrag[ct])[1] = *(const v8bf*)(p + 16); // next 8 K elems
            }
#pragma unroll
            for (int rt = 0; rt < 2; ++rt)
#pragma unroll
                for (int ct = 0; ct < 2; ++ct)
                    acc[rt][ct] = __builtin_amdgcn_wmma_f32_16x16x32_bf16(
                        false, af[rt], false, bfrag[ct], (short)0, acc[rt][ct], false, false);
        }
        if (s + 1 < nslab && issuer) __builtin_amdgcn_s_wait_tensorcnt(0);
        __syncthreads();
    }

    // C/D layout: VGPR r -> M = r (lanes 0-15) / r+8 (lanes 16-31); N = lane&15
#pragma unroll
    for (int rt = 0; rt < 2; ++rt) {
        int mb = bm + wm * 32 + rt * 16 + ((lane < 16) ? 0 : 8);
#pragma unroll
        for (int ct = 0; ct < 2; ++ct) {
            int n = bn + wn * 32 + ct * 16 + l15;
            float bv = bias ? bias[n] : 0.f;
#pragma unroll
            for (int r = 0; r < 8; ++r) {
                int m = mb + r;
                if (m < M) {
                    float v = acc[rt][ct][r] + bv;
                    size_t offc = (size_t)m * N + n;
                    if (OMODE == 0)      ((bf16*)Cout)[offc]  = (bf16)v;
                    else if (OMODE == 1) ((float*)Cout)[offc] = v;
                    else                 ((float*)Cout)[offc] = v + resid[offc];
                }
            }
        }
    }
}

// ---------------------------------------------------------------------------
// Windowed MHA, window=3, H=6, hd=128. One wave32 per (b,w,h); lane owns 4 dims.
__global__ void __launch_bounds__(128)
attn3_kernel(const bf16* __restrict__ qp, const bf16* __restrict__ kp,
             const bf16* __restrict__ vp, bf16* __restrict__ o,
             int Wn, int batchStride) {
    int gw = blockIdx.x * (blockDim.x >> 5) + (threadIdx.x >> 5);
    int lane = threadIdx.x & 31;
    int total = BB * Wn * HH;
    if (gw >= total) return;
    int h = gw % HH;
    int t = gw / HH;
    int w = t % Wn;
    int b = t / Wn;
    size_t base = (size_t)b * batchStride * DD + (size_t)(w * 3) * DD + h * HD + lane * 4;

    float q[3][4], k[3][4], v[3][4];
#pragma unroll
    for (int i = 0; i < 3; ++i)
#pragma unroll
        for (int j = 0; j < 4; ++j) {
            q[i][j] = (float)qp[base + i * DD + j];
            k[i][j] = (float)kp[base + i * DD + j];
            v[i][j] = (float)vp[base + i * DD + j];
        }
    float s[3][3];
#pragma unroll
    for (int i = 0; i < 3; ++i)
#pragma unroll
        for (int jj = 0; jj < 3; ++jj) {
            float p = 0.f;
#pragma unroll
            for (int d = 0; d < 4; ++d) p += q[i][d] * k[jj][d];
            for (int off = 16; off > 0; off >>= 1) p += __shfl_xor(p, off, 32);
            s[i][jj] = p * 0.08838834764831845f;  // 1/sqrt(128)
        }
#pragma unroll
    for (int i = 0; i < 3; ++i) {
        float mx = fmaxf(s[i][0], fmaxf(s[i][1], s[i][2]));
        float e0 = __expf(s[i][0] - mx), e1 = __expf(s[i][1] - mx), e2 = __expf(s[i][2] - mx);
        float inv = 1.f / (e0 + e1 + e2);
#pragma unroll
        for (int d = 0; d < 4; ++d) {
            float ov = (e0 * v[0][d] + e1 * v[1][d] + e2 * v[2][d]) * inv;
            o[base + i * DD + d] = (bf16)ov;
        }
    }
}

// ---------------------------------------------------------------------------
__global__ void upsample_gelu_acc(const bf16* __restrict__ so, float* __restrict__ acc,
                                  int Wn, int batchStride) {
    long i = (long)blockIdx.x * blockDim.x + threadIdx.x;
    const long total = (long)BB * TT * DD;
    if (i >= total) return;
    int d = (int)(i % DD);
    long r = i / DD;
    int t = (int)(r % TT);
    int b = (int)(r / TT);
    int WP = Wn * 3;
    int idx = (int)(((long)t * WP) / TT);
    float xv = (float)so[(size_t)b * batchStride * DD + (size_t)idx * DD + d];
    float gp = 0.5f * xv * (1.f + erff(xv * 0.7071067811865476f));
    acc[i] += gp;
}

// ---------------------------------------------------------------------------
extern "C" void kernel_launch(void* const* d_in, const int* in_sizes, int n_in,
                              void* d_out, int out_size, void* d_ws, size_t ws_size,
                              hipStream_t stream) {
    (void)in_sizes; (void)n_in; (void)out_size; (void)ws_size;
    const float* x   = (const float*)d_in[0];
    const float* lng = (const float*)d_in[1];
    const float* lnb = (const float*)d_in[2];
    const float* qw  = (const float*)d_in[3];
    const float* qb  = (const float*)d_in[4];
    const float* kw  = (const float*)d_in[5];
    const float* kb  = (const float*)d_in[6];
    const float* vw  = (const float*)d_in[7];
    const float* vb  = (const float*)d_in[8];
    const float* inw = (const float*)d_in[9];
    const float* inb = (const float*)d_in[10];
    const float* mow = (const float*)d_in[11];
    const float* mob = (const float*)d_in[12];
    const float* ow  = (const float*)d_in[13];
    const float* obv = (const float*)d_in[14];
    float* out = (float*)d_out;

    const size_t D = 768, MT = (size_t)BB * TT, szW = D * D;
    char* ws = (char*)d_ws;
    size_t off = 0;
    auto take = [&](size_t bytes) -> char* {
        off = (off + 255) & ~(size_t)255;
        char* p = ws + off;
        off += bytes;
        return p;
    };
    bf16* qwb  = (bf16*)take(szW * 2);
    bf16* kwb  = (bf16*)take(szW * 2);
    bf16* vwb  = (bf16*)take(szW * 2);
    bf16* inwb = (bf16*)take(3 * szW * 2);
    bf16* mowb = (bf16*)take(szW * 2);
    bf16* owb  = (bf16*)take(szW * 2);
    bf16* xnb  = (bf16*)take(MT * D * 2);  // reused later for bf16(acc)
    bf16* Qb   = (bf16*)take(MT * D * 2);
    bf16* Kb   = (bf16*)take(MT * D * 2);
    bf16* Vb   = (bf16*)take(MT * D * 2);
    bf16* pQ1  = (bf16*)take((size_t)BB * 1365 * D * 2);
    bf16* pK1  = (bf16*)take((size_t)BB * 1365 * D * 2);
    bf16* pV1  = (bf16*)take((size_t)BB * 1365 * D * 2);
    bf16* pQ2  = (bf16*)take((size_t)BB * 455 * D * 2);
    bf16* pK2  = (bf16*)take((size_t)BB * 455 * D * 2);
    bf16* pV2  = (bf16*)take((size_t)BB * 455 * D * 2);
    bf16* pQ3  = (bf16*)take((size_t)BB * 151 * D * 2);
    bf16* pK3  = (bf16*)take((size_t)BB * 151 * D * 2);
    bf16* pV3  = (bf16*)take((size_t)BB * 151 * D * 2);
    bf16* qp   = (bf16*)take(MT * D * 2);
    bf16* kpb  = (bf16*)take(MT * D * 2);
    bf16* vpb  = (bf16*)take(MT * D * 2);
    bf16* oat  = (bf16*)take(MT * D * 2);
    bf16* sob  = (bf16*)take(MT * D * 2);
    float* accf = (float*)take(MT * D * 4);

    auto cvt = [&](const float* src, bf16* dst, size_t n) {
        cvt_f32_bf16<<<dim3((unsigned)((n + 255) / 256)), 256, 0, stream>>>(src, dst, (long)n);
    };
    cvt(qw, qwb, szW); cvt(kw, kwb, szW); cvt(vw, vwb, szW);
    cvt(inw, inwb, 3 * szW); cvt(mow, mowb, szW); cvt(ow, owb, szW);

    ln_kernel<<<dim3((unsigned)MT), 256, 0, stream>>>(x, lng, lnb, xnb);

    auto gemm_bf = [&](const bf16* A, const bf16* Wm, const float* bias, bf16* C, int M) {
        gemm_wmma<0><<<dim3((unsigned)((M + 127) / 128), 12), 256, 0, stream>>>(
            A, Wm, bias, (void*)C, nullptr, M, (int)D, (int)D);
    };
    gemm_bf(xnb, qwb, qb, Qb, (int)MT);
    gemm_bf(xnb, kwb, kb, Kb, (int)MT);
    gemm_bf(xnb, vwb, vb, Vb, (int)MT);

    auto pool = [&](const bf16* in, bf16* outp, int inStride, int outLp) {
        long total = (long)BB * outLp * DD;
        pool3_kernel<<<dim3((unsigned)((total + 255) / 256)), 256, 0, stream>>>(in, outp, inStride, outLp);
    };
    pool(Qb, pQ1, 4096, 1365); pool(Kb, pK1, 4096, 1365); pool(Vb, pV1, 4096, 1365);
    pool(pQ1, pQ2, 1365, 455); pool(pK1, pK2, 1365, 455); pool(pV1, pV2, 1365, 455);
    pool(pQ2, pQ3, 455, 151);  pool(pK2, pK3, 455, 151);  pool(pV2, pV3, 455, 151);

    zero_f32<<<dim3((unsigned)((MT * D + 255) / 256)), 256, 0, stream>>>(accf, (long)(MT * D));

    struct LvlT { const bf16 *q, *k, *v; int Lp, W; };
    LvlT lv[4] = { {Qb, Kb, Vb, 4096, 1365},
                   {pQ1, pK1, pV1, 1365, 455},
                   {pQ2, pK2, pV2, 455, 151},
                   {pQ3, pK3, pV3, 151, 50} };
    for (int l = 0; l < 4; ++l) {
        int M = BB * lv[l].Lp;
        gemm_bf(lv[l].q, inwb,            inb,        qp,  M);
        gemm_bf(lv[l].k, inwb + szW,      inb + 768,  kpb, M);
        gemm_bf(lv[l].v, inwb + 2 * szW,  inb + 1536, vpb, M);
        int waves = BB * lv[l].W * HH;
        attn3_kernel<<<dim3((unsigned)((waves + 3) / 4)), 128, 0, stream>>>(
            qp, kpb, vpb, oat, lv[l].W, lv[l].Lp);
        gemm_bf(oat, mowb, mob, sob, M);
        long te = (long)BB * TT * DD;
        upsample_gelu_acc<<<dim3((unsigned)((te + 255) / 256)), 256, 0, stream>>>(
            sob, accf, lv[l].W, lv[l].Lp);
    }

    cvt(accf, xnb, MT * D);  // bf16(acc)
    gemm_wmma<2><<<dim3((unsigned)(MT / 128), 12), 256, 0, stream>>>(
        xnb, owb, obv, (void*)out, x, (int)MT, (int)D, (int)D);
}
